// DNDN_61830349193593
// MI455X (gfx1250) — matrix operand order repaired and checked by reading, not compiled, not run on hardware
//
#include <hip/hip_runtime.h>
#include <math.h>

#define TPB 256
#define HD 32               // hidden dim
#define NEG_SLOPE 0.2f
#define PRELU_W   0.1f
#define ENC_NEG_INF 0x007FFFFFu   // order-encoded -inf

typedef __attribute__((ext_vector_type(16))) _Float16 v16h;
typedef __attribute__((ext_vector_type(8)))  float    v8f;

// ---- order-preserving float<->uint encoding for atomicMax on floats ----
__device__ __forceinline__ unsigned f32_ord(float f) {
  unsigned u = __float_as_uint(f);
  return (u & 0x80000000u) ? ~u : (u | 0x80000000u);
}
__device__ __forceinline__ float ord_f32(unsigned u) {
  unsigned v = (u & 0x80000000u) ? (u & 0x7FFFFFFFu) : ~u;
  return __uint_as_float(v);
}

// =====================================================================
// Input layer: h = x * W_row  (x is [N,1], W is [1,32]); also attention
// dots el = h.a1, er = h.a2 for both src and snk paths.
// =====================================================================
__global__ __launch_bounds__(TPB)
void k_input_layer(const float* __restrict__ x,
                   const float* __restrict__ Ws, const float* __restrict__ a1s, const float* __restrict__ a2s,
                   const float* __restrict__ Wk, const float* __restrict__ a1k, const float* __restrict__ a2k,
                   float* __restrict__ hs, float* __restrict__ hk,
                   float* __restrict__ el_s, float* __restrict__ er_s,
                   float* __restrict__ el_k, float* __restrict__ er_k, int N) {
  int n = blockIdx.x * blockDim.x + threadIdx.x;
  if (n >= N) return;
  float xv = x[n];
  float es = 0.f, rs = 0.f, ek = 0.f, rk = 0.f;
#pragma unroll
  for (int j = 0; j < HD; ++j) {
    float vs = xv * Ws[j];
    float vk = xv * Wk[j];
    hs[n * HD + j] = vs;
    hk[n * HD + j] = vk;
    es += vs * a1s[j]; rs += vs * a2s[j];
    ek += vk * a1k[j]; rk += vk * a2k[j];
  }
  el_s[n] = es; er_s[n] = rs; el_k[n] = ek; er_k[n] = rk;
}

// =====================================================================
// GAT aggregation pipeline (4 kernels + prelu)
// =====================================================================
__global__ __launch_bounds__(TPB)
void k_gat_init(const float* __restrict__ bias, float* __restrict__ out,
                unsigned* __restrict__ maxb, float* __restrict__ denom, int N) {
  int tid = blockIdx.x * blockDim.x + threadIdx.x;
  if (tid < N) { maxb[tid] = ENC_NEG_INF; denom[tid] = 0.f; }
  if (tid < N * HD) out[tid] = bias[tid & (HD - 1)];
}

__global__ __launch_bounds__(TPB)
void k_edge_logit_max(const int* __restrict__ ei, const float* __restrict__ el,
                      const float* __restrict__ er, float* __restrict__ ebuf,
                      unsigned* __restrict__ maxb, int E) {
  int t = blockIdx.x * blockDim.x + threadIdx.x;
  if (t >= E) return;
  int s = ei[t], d = ei[E + t];
  float e = el[s] + er[d];
  e = (e >= 0.f) ? e : NEG_SLOPE * e;     // leaky_relu
  ebuf[t] = e;
  atomicMax(&maxb[d], f32_ord(e));
}

__global__ __launch_bounds__(TPB)
void k_edge_exp(const int* __restrict__ ei, float* __restrict__ ebuf,
                const unsigned* __restrict__ maxb, float* __restrict__ denom, int E) {
  int t = blockIdx.x * blockDim.x + threadIdx.x;
  if (t >= E) return;
  int d = ei[E + t];
  float m = ord_f32(maxb[d]);
  if (!(m >= -3.4e38f && m <= 3.4e38f)) m = 0.f;   // non-finite -> 0
  float p = expf(ebuf[t] - m);
  ebuf[t] = p;
  atomicAdd(&denom[d], p);
}

// one lane per (edge, feature): 32 lanes of a wave share one edge -> e/denom
// loads broadcast, h[s] gather and out[d] atomics fully coalesced (128B).
__global__ __launch_bounds__(TPB)
void k_edge_aggregate(const int* __restrict__ ei, const float* __restrict__ ebuf,
                      const float* __restrict__ denom, const float* __restrict__ h,
                      float* __restrict__ out, int E) {
  int tid = blockIdx.x * blockDim.x + threadIdx.x;
  int t = tid >> 5;
  int j = tid & 31;
  if (t >= E) return;
  int s = ei[t], d = ei[E + t];
  float alpha = ebuf[t] / (denom[d] + 1e-16f);
  atomicAdd(&out[d * HD + j], alpha * h[s * HD + j]);
}

__global__ __launch_bounds__(TPB)
void k_prelu(float* __restrict__ buf, int n) {
  int i = blockIdx.x * blockDim.x + threadIdx.x;
  if (i >= n) return;
  float v = buf[i];
  buf[i] = (v >= 0.f) ? v : PRELU_W * v;
}

__global__ __launch_bounds__(TPB)
void k_combine(const float* __restrict__ a, const float* __restrict__ b,
               float* __restrict__ c, int n) {
  int i = blockIdx.x * blockDim.x + threadIdx.x;
  if (i < n) c[i] = a[i] + b[i];
}

// =====================================================================
// Dual dense linear via WMMA: ha = c @ Wa, hb = c @ Wb  (c:[N,32], W:[32,32])
// One wave per 16-row tile; A reused across 4 v_wmma_f32_16x16x32_f16.
// Both weight matrices are staged transposed+f16 in LDS so each lane's
// B fragment is one aligned 32B LDS read (ds_load_b128 x2).
// A as loaded here (contiguous k, not the ISA table order): lanes 0-15 row M=lane, K=0..15;
//                          lanes 16-31 row M=lane-16, K=16..31.
// B as loaded here (same contiguous k as A): lanes 0-15 col N=lane, K=0..15;
//                          lanes 16-31 col N=lane-16, K=16..31.
// D layout (f32 16x16): lanes 0-15 N=lane, VGPR j = row j; lanes16-31 row 8+j.
// =====================================================================
__global__ __launch_bounds__(TPB)
void k_linear2_wmma(const float* __restrict__ c,
                    const float* __restrict__ Wa, const float* __restrict__ Wb,
                    float* __restrict__ ha, float* __restrict__ hb,
                    int N, int nTiles) {
  __shared__ __align__(64) _Float16 WtA[HD][HD];  // WtA[col][k] = Wa[k][col]
  __shared__ __align__(64) _Float16 WtB[HD][HD];
  // cooperative transpose + f32->f16 convert (coalesced global reads)
  for (int e = threadIdx.x; e < HD * HD; e += TPB) {
    int k = e >> 5, col = e & (HD - 1);
    WtA[col][k] = (_Float16)Wa[e];
    WtB[col][k] = (_Float16)Wb[e];
  }
  __syncthreads();

  int wave = blockIdx.x * (blockDim.x >> 5) + (threadIdx.x >> 5);
  if (wave >= nTiles) return;          // wave-uniform guard: EXEC stays all-ones
  int lane = threadIdx.x & 31;
  int row0 = wave << 4;
  int mrow = row0 + (lane & 15);
  int rc = (mrow < N) ? mrow : (N - 1);          // clamp tail loads

  // ---- A fragment: one contiguous 16-float run -> 4x float4 loads ----
  const float4* crow4 = (const float4*)(c + (size_t)rc * HD);
  int i0 = (lane < 16) ? 0 : 4;                  // kA/4
  float4 a0 = crow4[i0];
  float4 a1 = crow4[i0 + 1];
  float4 a2 = crow4[i0 + 2];
  float4 a3 = crow4[i0 + 3];
  v16h A;
  A[0]  = (_Float16)a0.x; A[1]  = (_Float16)a0.y; A[2]  = (_Float16)a0.z; A[3]  = (_Float16)a0.w;
  A[4]  = (_Float16)a1.x; A[5]  = (_Float16)a1.y; A[6]  = (_Float16)a1.z; A[7]  = (_Float16)a1.w;
  A[8]  = (_Float16)a2.x; A[9]  = (_Float16)a2.y; A[10] = (_Float16)a2.z; A[11] = (_Float16)a2.w;
  A[12] = (_Float16)a3.x; A[13] = (_Float16)a3.y; A[14] = (_Float16)a3.z; A[15] = (_Float16)a3.w;

  // ---- B fragments: one aligned 32B LDS read each ----
  int col = lane & 15;
  int kB = (lane < 16) ? 0 : 16;
  v16h Ba0 = *(const v16h*)&WtA[col][kB];
  v16h Ba1 = *(const v16h*)&WtA[col + 16][kB];
  v16h Bb0 = *(const v16h*)&WtB[col][kB];
  v16h Bb1 = *(const v16h*)&WtB[col + 16][kB];

  v8f z = {0.f, 0.f, 0.f, 0.f, 0.f, 0.f, 0.f, 0.f};
  v8f d0 = __builtin_amdgcn_wmma_f32_16x16x32_f16(false, A, false, Ba0, (short)0, z, false, false);
  v8f d1 = __builtin_amdgcn_wmma_f32_16x16x32_f16(false, A, false, Ba1, (short)0, z, false, false);
  v8f d2 = __builtin_amdgcn_wmma_f32_16x16x32_f16(false, A, false, Bb0, (short)0, z, false, false);
  v8f d3 = __builtin_amdgcn_wmma_f32_16x16x32_f16(false, A, false, Bb1, (short)0, z, false, false);

  int mb = (lane < 16) ? 0 : 8;
  float* pa = ha + (size_t)(row0 + mb) * HD + col;
  float* pb = hb + (size_t)(row0 + mb) * HD + col;
  if (row0 + 16 <= N) {                // wave-uniform fast path: no EXEC churn
#pragma unroll
    for (int j = 0; j < 8; ++j) {
      pa[(size_t)j * HD]      = d0[j];
      pa[(size_t)j * HD + 16] = d1[j];
      pb[(size_t)j * HD]      = d2[j];
      pb[(size_t)j * HD + 16] = d3[j];
    }
  } else {                             // tail tile (never taken when 16 | N)
#pragma unroll
    for (int j = 0; j < 8; ++j) {
      if (row0 + mb + j < N) {
        pa[(size_t)j * HD]      = d0[j];
        pa[(size_t)j * HD + 16] = d1[j];
        pb[(size_t)j * HD]      = d2[j];
        pb[(size_t)j * HD + 16] = d3[j];
      }
    }
  }
}

// attention dot products for hidden layers
__global__ __launch_bounds__(TPB)
void k_att_dots(const float* __restrict__ hs, const float* __restrict__ hk,
                const float* __restrict__ a1s, const float* __restrict__ a2s,
                const float* __restrict__ a1k, const float* __restrict__ a2k,
                float* __restrict__ el_s, float* __restrict__ er_s,
                float* __restrict__ el_k, float* __restrict__ er_k, int N) {
  int n = blockIdx.x * blockDim.x + threadIdx.x;
  if (n >= N) return;
  float es = 0.f, rs = 0.f, ek = 0.f, rk = 0.f;
#pragma unroll
  for (int j = 0; j < HD; ++j) {
    float vs = hs[n * HD + j];
    float vk = hk[n * HD + j];
    es += vs * a1s[j]; rs += vs * a2s[j];
    ek += vk * a1k[j]; rk += vk * a2k[j];
  }
  el_s[n] = es; er_s[n] = rs; el_k[n] = ek; er_k[n] = rk;
}

// =====================================================================
// Output heads
// =====================================================================
__global__ __launch_bounds__(TPB)
void k_final_dim0(const float* __restrict__ se, const float* __restrict__ sk,
                  const float* __restrict__ Wos, const float* __restrict__ bos,
                  const float* __restrict__ Wok, const float* __restrict__ bok,
                  float* __restrict__ out, int N) {
  int n = blockIdx.x * blockDim.x + threadIdx.x;
  if (n >= N) return;
  float s0 = bos[0], s1 = bos[1], q0 = bok[0], q1 = bok[1];
#pragma unroll
  for (int k = 0; k < 2 * HD; ++k) {
    float v = (k < HD) ? se[n * HD + k] : sk[n * HD + k - HD];
    s0 += v * Wos[k * 2];  s1 += v * Wos[k * 2 + 1];
    q0 += v * Wok[k * 2];  q1 += v * Wok[k * 2 + 1];
  }
  s0 = fmaxf(s0, 0.f); s1 = fmaxf(s1, 0.f);
  q0 = fmaxf(q0, 0.f); q1 = fmaxf(q1, 0.f);
  out[n * 4 + 0] = 0.5f * (s0 + q0);
  out[n * 4 + 1] = 0.5f * (s1 + q1);
}

// agg = comb/2  (the "+ emb/2" term of _edge_neighbor)
__global__ __launch_bounds__(TPB)
void k_agg64_init(const float* __restrict__ se, const float* __restrict__ sk,
                  float* __restrict__ agg, int N) {
  int tid = blockIdx.x * blockDim.x + threadIdx.x;
  if (tid >= N * 2 * HD) return;
  int k = tid & (2 * HD - 1);
  int n = tid >> 6;
  float v = (k < HD) ? se[n * HD + k] : sk[n * HD + k - HD];
  agg[tid] = 0.5f * v;
}

// one lane per (edge, feature 0..63)
__global__ __launch_bounds__(TPB)
void k_edge_scatter64(const int* __restrict__ ei, const float* __restrict__ se,
                      const float* __restrict__ sk, const float* __restrict__ x,
                      float* __restrict__ agg, int E) {
  int tid = blockIdx.x * blockDim.x + threadIdx.x;
  int t = tid >> 6;
  int k = tid & 63;
  if (t >= E) return;
  int s = ei[t], d = ei[E + t];
  float xv = x[s];
  float v = ((k < HD) ? se[s * HD + k] : sk[s * HD + k - HD]) * xv;
  atomicAdd(&agg[d * 2 * HD + k], v);
}

__global__ __launch_bounds__(TPB)
void k_relu_copy(const float* __restrict__ src, float* __restrict__ dst, int n) {
  int i = blockIdx.x * blockDim.x + threadIdx.x;
  if (i < n) dst[i] = fmaxf(src[i], 0.f);
}

__global__ __launch_bounds__(TPB)
void k_final_dim1(const float* __restrict__ agg, const float* __restrict__ sum64,
                  const float* __restrict__ Wd1, const float* __restrict__ bd1,
                  float* __restrict__ out, int N) {
  int n = blockIdx.x * blockDim.x + threadIdx.x;
  if (n >= N) return;
  float v0 = bd1[0], v1 = bd1[1];
#pragma unroll
  for (int k = 0; k < 2 * HD; ++k) {
    float v = fmaxf(agg[n * 2 * HD + k], 0.f) + sum64[n * 2 * HD + k];
    v0 += v * Wd1[k * 2];
    v1 += v * Wd1[k * 2 + 1];
  }
  out[n * 4 + 2] = fmaxf(v0, 0.f);
  out[n * 4 + 3] = fmaxf(v1, 0.f);
}

// =====================================================================
// Host orchestration
// =====================================================================
static inline int cdiv(long a, long b) { return (int)((a + b - 1) / b); }

static void run_gat(hipStream_t stream, int N, int E, const int* ei,
                    const float* h, const float* el, const float* er,
                    const float* bias, float* out,
                    unsigned* maxb, float* denom, float* ebuf) {
  int gN32 = cdiv((long)N * HD, TPB);
  int gE   = cdiv(E, TPB);
  int gE32 = cdiv((long)E * 32, TPB);
  k_gat_init<<<gN32, TPB, 0, stream>>>(bias, out, maxb, denom, N);
  k_edge_logit_max<<<gE, TPB, 0, stream>>>(ei, el, er, ebuf, maxb, E);
  k_edge_exp<<<gE, TPB, 0, stream>>>(ei, ebuf, maxb, denom, E);
  k_edge_aggregate<<<gE32, TPB, 0, stream>>>(ei, ebuf, denom, h, out, E);
  k_prelu<<<gN32, TPB, 0, stream>>>(out, N * HD);
}

extern "C" void kernel_launch(void* const* d_in, const int* in_sizes, int n_in,
                              void* d_out, int out_size, void* d_ws, size_t ws_size,
                              hipStream_t stream) {
  const float* x        = (const float*)d_in[0];
  const int*   ei_src   = (const int*)d_in[1];
  const int*   ei_snk   = (const int*)d_in[2];
  const float* W_in_src = (const float*)d_in[3];
  const float* b_in_src = (const float*)d_in[4];
  const float* a1_in_src= (const float*)d_in[5];
  const float* a2_in_src= (const float*)d_in[6];
  const float* W_in_snk = (const float*)d_in[7];
  const float* b_in_snk = (const float*)d_in[8];
  const float* a1_in_snk= (const float*)d_in[9];
  const float* a2_in_snk= (const float*)d_in[10];
  const float* W_src    = (const float*)d_in[11];
  const float* b_src    = (const float*)d_in[12];
  const float* a1_src   = (const float*)d_in[13];
  const float* a2_src   = (const float*)d_in[14];
  const float* W_snk    = (const float*)d_in[15];
  const float* b_snk    = (const float*)d_in[16];
  const float* a1_snk   = (const float*)d_in[17];
  const float* a2_snk   = (const float*)d_in[18];
  const float* W_o_src  = (const float*)d_in[19];
  const float* b_o_src  = (const float*)d_in[20];
  const float* W_o_snk  = (const float*)d_in[21];
  const float* b_o_snk  = (const float*)d_in[22];
  const float* W_d1     = (const float*)d_in[23];
  const float* b_d1     = (const float*)d_in[24];
  float* out = (float*)d_out;

  const int N = in_sizes[0];
  const int E = in_sizes[1] / 2;
  const int hidden_layers = in_sizes[11] / (HD * HD);   // L-1

  // ---- workspace layout (floats) ----
  float* ws = (float*)d_ws;
  float* se   = ws;               ws += (size_t)N * HD;
  float* sk   = ws;               ws += (size_t)N * HD;
  float* cbuf = ws;               ws += (size_t)N * HD;
  float* hs   = ws;               ws += (size_t)N * HD;
  float* hk   = ws;               ws += (size_t)N * HD;
  float* el_s = ws;               ws += N;
  float* er_s = ws;               ws += N;
  float* el_k = ws;               ws += N;
  float* er_k = ws;               ws += N;
  float* denom= ws;               ws += N;
  unsigned* maxb = (unsigned*)ws; ws += N;
  float* ebuf = ws;               ws += E;
  float* agg  = ws;               ws += (size_t)N * 2 * HD;
  float* sum64= ws;               ws += (size_t)N * 2 * HD;

  int gN   = cdiv(N, TPB);
  int gN32 = cdiv((long)N * HD, TPB);
  int gN64 = cdiv((long)N * 2 * HD, TPB);
  int gE64 = cdiv((long)E * 2 * HD, TPB);
  int nTiles = cdiv(N, 16);
  int wavesPerBlock = TPB / 32;
  int gTiles = cdiv(nTiles, wavesPerBlock);

  // ---- input GAT layer ----
  k_input_layer<<<gN, TPB, 0, stream>>>(x, W_in_src, a1_in_src, a2_in_src,
                                        W_in_snk, a1_in_snk, a2_in_snk,
                                        hs, hk, el_s, er_s, el_k, er_k, N);
  run_gat(stream, N, E, ei_src, hs, el_s, er_s, b_in_src, se, maxb, denom, ebuf);
  run_gat(stream, N, E, ei_snk, hk, el_k, er_k, b_in_snk, sk, maxb, denom, ebuf);

  // ---- hidden GAT layers ----
  for (int i = 0; i < hidden_layers; ++i) {
    k_combine<<<gN32, TPB, 0, stream>>>(se, sk, cbuf, N * HD);
    k_linear2_wmma<<<gTiles, TPB, 0, stream>>>(cbuf,
                                               W_src + (size_t)i * HD * HD,
                                               W_snk + (size_t)i * HD * HD,
                                               hs, hk, N, nTiles);
    k_att_dots<<<gN, TPB, 0, stream>>>(hs, hk,
                                       a1_src + i * HD, a2_src + i * HD,
                                       a1_snk + i * HD, a2_snk + i * HD,
                                       el_s, er_s, el_k, er_k, N);
    run_gat(stream, N, E, ei_src, hs, el_s, er_s, b_src + i * HD, se, maxb, denom, ebuf);
    run_gat(stream, N, E, ei_snk, hk, el_k, er_k, b_snk + i * HD, sk, maxb, denom, ebuf);
  }

  // ---- output dim 0/1 ----
  k_final_dim0<<<gN, TPB, 0, stream>>>(se, sk, W_o_src, b_o_src, W_o_snk, b_o_snk, out, N);

  // ---- edge_neighbor heads: relu(agg_src) + relu(agg_snk) -> dim 2/3 ----
  k_agg64_init<<<gN64, TPB, 0, stream>>>(se, sk, agg, N);
  k_edge_scatter64<<<gE64, TPB, 0, stream>>>(ei_src, se, sk, x, agg, E);
  k_relu_copy<<<gN64, TPB, 0, stream>>>(agg, sum64, N * 2 * HD);
  k_agg64_init<<<gN64, TPB, 0, stream>>>(se, sk, agg, N);
  k_edge_scatter64<<<gE64, TPB, 0, stream>>>(ei_snk, se, sk, x, agg, E);
  k_final_dim1<<<gN, TPB, 0, stream>>>(agg, sum64, W_d1, b_d1, out, N);
}
